// FPModule_33217277067475
// MI455X (gfx1250) — matrix operand order repaired and checked by reading, not compiled, not run on hardware
//
#include <hip/hip_runtime.h>
#include <hip/hip_bf16.h>

typedef __attribute__((ext_vector_type(16))) __bf16 v16bf;
typedef __attribute__((ext_vector_type(8)))  __bf16 v8bf;
typedef __attribute__((ext_vector_type(8)))  float  v8f;

#define NB   16
#define NS   1024
#define NT   4096
#define CIN  256
#define CSK  64
#define HID  256
#define HIN  320            // CIN + CSK
#define NQ   (NB * NT)      // 65536 query rows

static __device__ inline v16bf cat8(v8bf lo, v8bf hi) {
    return __builtin_shufflevector(lo, hi, 0,1,2,3,4,5,6,7,8,9,10,11,12,13,14,15);
}

// ---------------------------------------------------------------------------
// 1) Weight convert + transpose: W1[320,256]->W1t bf16[256,320],
//    W2[256,256]->W2t bf16[256,256]
// ---------------------------------------------------------------------------
__global__ void conv_weights(const float* __restrict__ W1, const float* __restrict__ W2,
                             __bf16* __restrict__ W1t, __bf16* __restrict__ W2t) {
    int i = blockIdx.x * blockDim.x + threadIdx.x;
    const int n1 = HIN * HID;           // 81920
    const int n2 = HID * HID;           // 65536
    if (i < n1) {
        int k = i / HID, n = i % HID;
        W1t[n * HIN + k] = (__bf16)W1[i];
    } else if (i < n1 + n2) {
        int j = i - n1;
        int k = j / HID, n = j % HID;
        W2t[n * HID + k] = (__bf16)W2[j];
    }
}

// ---------------------------------------------------------------------------
// 2) kNN(K=3): one block = 256 queries of one batch; coarse positions in LDS
// ---------------------------------------------------------------------------
__global__ void knn_kernel(const float* __restrict__ pos,
                           const float* __restrict__ pos_skip,
                           int* __restrict__ idx_out,
                           float* __restrict__ w_out) {
    __shared__ float sx[NS], sy[NS], sz[NS];
    const int tid   = threadIdx.x;
    const int batch = blockIdx.x >> 4;      // 16 blocks per batch
    const int blkq  = blockIdx.x & 15;

    for (int t = 0; t < NS / 256; ++t) {
        int j = t * 256 + tid;
        const float* p = pos + (size_t)(batch * NS + j) * 3;
        sx[j] = p[0]; sy[j] = p[1]; sz[j] = p[2];
    }
    __syncthreads();

    const int q = batch * NT + blkq * 256 + tid;
    const float qx = pos_skip[(size_t)q * 3 + 0];
    const float qy = pos_skip[(size_t)q * 3 + 1];
    const float qz = pos_skip[(size_t)q * 3 + 2];

    float d0 = 3.0e38f, d1 = 3.0e38f, d2 = 3.0e38f;
    int   j0 = 0, j1 = 0, j2 = 0;
    #pragma unroll 4
    for (int j = 0; j < NS; ++j) {
        float dx = qx - sx[j], dy = qy - sy[j], dz = qz - sz[j];
        float d  = fmaf(dz, dz, fmaf(dy, dy, dx * dx));
        if (d < d2) {
            if (d < d1) {
                d2 = d1; j2 = j1;
                if (d < d0) { d1 = d0; j1 = j0; d0 = d; j0 = j; }
                else        { d1 = d;  j1 = j; }
            } else          { d2 = d;  j2 = j; }
        }
    }
    float w0 = 1.0f / fmaxf(d0, 1e-16f);
    float w1 = 1.0f / fmaxf(d1, 1e-16f);
    float w2 = 1.0f / fmaxf(d2, 1e-16f);
    float inv = 1.0f / (w0 + w1 + w2);
    idx_out[q * 3 + 0] = batch * NS + j0;
    idx_out[q * 3 + 1] = batch * NS + j1;
    idx_out[q * 3 + 2] = batch * NS + j2;
    w_out[q * 3 + 0] = w0 * inv;
    w_out[q * 3 + 1] = w1 * inv;
    w_out[q * 3 + 2] = w2 * inv;
}

// ---------------------------------------------------------------------------
// 3) Interpolate + concat skip -> hin bf16 [NQ, 320]
// ---------------------------------------------------------------------------
__global__ void interp_kernel(const float* __restrict__ x,
                              const float* __restrict__ x_skip,
                              const int* __restrict__ idx,
                              const float* __restrict__ w,
                              __bf16* __restrict__ hin) {
    const int q = blockIdx.x;
    const int c = threadIdx.x;          // blockDim = 320
    if (c < CIN) {
        const int  i0 = idx[q * 3 + 0], i1 = idx[q * 3 + 1], i2 = idx[q * 3 + 2];
        const float w0 = w[q * 3 + 0], w1 = w[q * 3 + 1], w2 = w[q * 3 + 2];
        float y = w0 * x[(size_t)i0 * CIN + c]
                + w1 * x[(size_t)i1 * CIN + c]
                + w2 * x[(size_t)i2 * CIN + c];
        hin[(size_t)q * HIN + c] = (__bf16)y;
    } else {
        hin[(size_t)q * HIN + c] = (__bf16)x_skip[(size_t)q * CSK + (c - CIN)];
    }
}

// ---------------------------------------------------------------------------
// 4) WMMA GEMM: out[M,256] = relu(A[M,Ktot] @ Wt^T + bias)
//    A bf16 row-major [M,Ktot]; Wt bf16 [256,Ktot] (transposed weights).
//    One wave computes a 32(M) x 16(N) tile (two accumulators share B frag).
// ---------------------------------------------------------------------------
template <bool F32OUT>
__global__ void gemm_wmma(const __bf16* __restrict__ A,
                          const __bf16* __restrict__ Wt,
                          const float* __restrict__ bias,
                          void* __restrict__ out, int Ktot) {
    const int wave = (blockIdx.x * blockDim.x + threadIdx.x) >> 5;
    const int lane = threadIdx.x & 31;
    const int tn   = wave & 15;         // 16 N-tiles (256/16)
    const int tm2  = wave >> 4;         // each covers 32 rows

    const int lrow = lane & 15;
    const int half = lane >> 4;         // 0 or 1
    const size_t row0 = (size_t)tm2 * 32 + lrow;
    const size_t row1 = row0 + 16;
    const int ncol = tn * 16 + lrow;
    const int kh   = half * 8;          // A: K sub-offset
    const int kbh  = half * 16;         // unused: B uses kh, same order as A

    const __bf16* a0p = A + row0 * Ktot;
    const __bf16* a1p = A + row1 * Ktot;
    const __bf16* bp  = Wt + (size_t)ncol * Ktot;

    v8f c0 = {};
    v8f c1 = {};
    #pragma unroll 2
    for (int kb = 0; kb < Ktot; kb += 32) {
        v8bf a0lo = *(const v8bf*)(a0p + kb + kh);
        v8bf a0hi = *(const v8bf*)(a0p + kb + 16 + kh);
        v8bf a1lo = *(const v8bf*)(a1p + kb + kh);
        v8bf a1hi = *(const v8bf*)(a1p + kb + 16 + kh);
        v16bf bfrag = cat8(*(const v8bf*)(bp + kb + kh), *(const v8bf*)(bp + kb + 16 + kh));
        v16bf a0f = cat8(a0lo, a0hi);
        v16bf a1f = cat8(a1lo, a1hi);
        c0 = __builtin_amdgcn_wmma_f32_16x16x32_bf16(false, a0f, false, bfrag,
                                                     (short)0, c0, false, false);
        c1 = __builtin_amdgcn_wmma_f32_16x16x32_bf16(false, a1f, false, bfrag,
                                                     (short)0, c1, false, false);
    }

    const float bv = bias[ncol];
    #pragma unroll
    for (int v = 0; v < 8; ++v) {
        const int m = v + half * 8;
        float x0 = fmaxf(c0[v] + bv, 0.0f);
        float x1 = fmaxf(c1[v] + bv, 0.0f);
        size_t r0 = (size_t)tm2 * 32 + m;
        size_t r1 = r0 + 16;
        if (F32OUT) {
            ((float*)out)[r0 * HID + ncol] = x0;
            ((float*)out)[r1 * HID + ncol] = x1;
        } else {
            ((__bf16*)out)[r0 * HID + ncol] = (__bf16)x0;
            ((__bf16*)out)[r1 * HID + ncol] = (__bf16)x1;
        }
    }
}

// ---------------------------------------------------------------------------
// 5) Tail: copy pos_skip (f32) and batch_skip (i64) into output tuple tail
// ---------------------------------------------------------------------------
__global__ void tail_copy(const float* __restrict__ pos_skip,
                          const long long* __restrict__ batch_skip,
                          float* __restrict__ out_pos,
                          long long* __restrict__ out_batch) {
    int i = blockIdx.x * blockDim.x + threadIdx.x;
    if (i < NQ * 3) out_pos[i] = pos_skip[i];
    if (i < NQ)     out_batch[i] = batch_skip[i];
}

// ---------------------------------------------------------------------------
extern "C" void kernel_launch(void* const* d_in, const int* in_sizes, int n_in,
                              void* d_out, int out_size, void* d_ws, size_t ws_size,
                              hipStream_t stream) {
    const float*     x         = (const float*)d_in[0];
    const float*     pos       = (const float*)d_in[1];
    const float*     x_skip    = (const float*)d_in[2];
    const float*     pos_skip  = (const float*)d_in[3];
    const float*     W1        = (const float*)d_in[4];
    const float*     b1        = (const float*)d_in[5];
    const float*     W2        = (const float*)d_in[6];
    const float*     b2        = (const float*)d_in[7];
    const long long* batch_sk  = (const long long*)d_in[9];

    // workspace layout (all offsets 1KB-multiple aligned)
    char* ws = (char*)d_ws;
    __bf16* hin = (__bf16*)ws;                         ws += (size_t)NQ * HIN * 2;  // 40 MB
    __bf16* h1  = (__bf16*)ws;                         ws += (size_t)NQ * HID * 2;  // 32 MB
    __bf16* W1t = (__bf16*)ws;                         ws += (size_t)HID * HIN * 2;
    __bf16* W2t = (__bf16*)ws;                         ws += (size_t)HID * HID * 2;
    int*    knn_idx = (int*)ws;                        ws += (size_t)NQ * 3 * 4;
    float*  knn_w   = (float*)ws;                      ws += (size_t)NQ * 3 * 4;

    float*     out_h     = (float*)d_out;
    float*     out_pos   = out_h + (size_t)NQ * HID;
    long long* out_batch = (long long*)(out_pos + (size_t)NQ * 3);

    // 1) weights -> bf16 transposed
    {
        int n = HIN * HID + HID * HID;
        conv_weights<<<(n + 255) / 256, 256, 0, stream>>>(W1, W2, W1t, W2t);
    }
    // 2) kNN
    knn_kernel<<<NB * (NT / 256), 256, 0, stream>>>(pos, pos_skip, knn_idx, knn_w);
    // 3) interpolate + concat (blockDim 320 = 10 waves)
    interp_kernel<<<NQ, HIN, 0, stream>>>(x, x_skip, knn_idx, knn_w, hin);
    // 4) GEMM1: hin[NQ,320] @ W1 -> h1 bf16
    {
        int waves = (NQ / 32) * 16;                    // 32768 waves
        gemm_wmma<false><<<waves * 32 / 256, 256, 0, stream>>>(hin, W1t, b1, h1, HIN);
    }
    // 5) GEMM2: h1[NQ,256] @ W2 -> out f32
    {
        int waves = (NQ / 32) * 16;
        gemm_wmma<true><<<waves * 32 / 256, 256, 0, stream>>>(h1, W2t, b2, out_h, HID);
    }
    // 6) tuple tail
    tail_copy<<<(NQ * 3 + 255) / 256, 256, 0, stream>>>(pos_skip, batch_sk, out_pos, out_batch);
}
